// GraphAttentionLayer_63831803953677
// MI455X (gfx1250) — matrix operand order repaired and checked by reading, not compiled, not run on hardware
//
#include <hip/hip_runtime.h>

#define N_NODES 8192
#define F_IN    768
#define F_OUT   128

typedef __attribute__((ext_vector_type(2)))  float     v2f;
typedef __attribute__((ext_vector_type(4)))  float     fx4;
typedef __attribute__((ext_vector_type(8)))  float     v8f;
typedef __attribute__((ext_vector_type(4)))  int       v4i;
typedef __attribute__((ext_vector_type(8)))  _Float16  v8h;
typedef __attribute__((ext_vector_type(16))) _Float16  v16h;

#ifndef __has_builtin
#define __has_builtin(x) 0
#endif
#if __has_builtin(__builtin_amdgcn_global_load_async_to_lds_b128) && \
    __has_builtin(__builtin_amdgcn_s_wait_asynccnt)
#define USE_ASYNC_LDS 1
#else
#define USE_ASYNC_LDS 0
#endif

// builtin expects typed vector pointers: AS1 (global) src, AS3 (LDS) dst
#define GASV(p) ((__attribute__((address_space(1))) v4i*)(p))
#define LASV(p) ((__attribute__((address_space(3))) v4i*)(p))

// ---------------------------------------------------------------------------
// Kernel A: Hp = H @ W  (fp32 WMMA 16x16x4), also emit HpT in f16 for stage 2.
// Grid: 512 blocks (16 rows each) x 256 threads (8 waves -> 8 feature tiles).
// ---------------------------------------------------------------------------
__global__ __launch_bounds__(256)
void gat_hp_gemm(const float* __restrict__ H, const float* __restrict__ W,
                 float* __restrict__ Hp, _Float16* __restrict__ HpT) {
    const int lane  = threadIdx.x & 31;
    const int wave  = threadIdx.x >> 5;        // feature tile 0..7
    const int ibase = blockIdx.x * 16;
    const int half  = lane >> 4;               // 0 | 1
    const int l16   = lane & 15;
    const int row   = ibase + l16;             // A row for this lane
    const int col   = wave * 16 + l16;         // B column for this lane

    v8f c = {};
    const float* hrow = H + (size_t)row * F_IN + 2 * half;
    for (int k = 0; k < F_IN; k += 4) {
        // A 16x4 fp32 layout: VGPR v holds K = k + 2*half + v
        v2f a = *(const v2f*)(hrow + k);
        // B 4x16 fp32 layout (mirror): VGPR v holds K = k + 2*half + v, N = col
        const int kr = k + 2 * half;
        v2f b;
        b[0] = W[(size_t)kr * F_OUT + col];
        b[1] = W[(size_t)(kr + 1) * F_OUT + col];
        c = __builtin_amdgcn_wmma_f32_16x16x4_f32(false, a, false, b,
                                                  (short)0, c, false, false);
    }
    // C layout: VGPR k -> row (ibase + k + 8*half), col
#pragma unroll
    for (int k = 0; k < 8; ++k) {
        const int r = ibase + k + 8 * half;
        Hp [(size_t)r * F_OUT + col]     = c[k];
        HpT[(size_t)col * N_NODES + r]   = (_Float16)c[k];
    }
}

// ---------------------------------------------------------------------------
// Kernel B: src = Hp @ a[:F], dst = Hp @ a[F:]   (one block per row)
// ---------------------------------------------------------------------------
__global__ __launch_bounds__(128)
void gat_src_dst(const float* __restrict__ Hp, const float* __restrict__ a,
                 float* __restrict__ src, float* __restrict__ dst) {
    __shared__ float s1[128];
    __shared__ float s2[128];
    const int row = blockIdx.x;
    const int f   = threadIdx.x;
    const float v = Hp[(size_t)row * F_OUT + f];
    s1[f] = v * a[f];
    s2[f] = v * a[F_OUT + f];
    __syncthreads();
    for (int s = 64; s > 0; s >>= 1) {
        if (f < s) { s1[f] += s1[f + s]; s2[f] += s2[f + s]; }
        __syncthreads();
    }
    if (f == 0) { src[row] = s1[0]; dst[row] = s2[0]; }
}

// ---------------------------------------------------------------------------
// Kernel C: dmax = max(dst)
// ---------------------------------------------------------------------------
__global__ __launch_bounds__(256)
void gat_dmax(const float* __restrict__ dst, float* __restrict__ dmax) {
    __shared__ float sm[256];
    float m = -3.4e38f;
    for (int i = threadIdx.x; i < N_NODES; i += 256) m = fmaxf(m, dst[i]);
    sm[threadIdx.x] = m;
    __syncthreads();
    for (int s = 128; s > 0; s >>= 1) {
        if (threadIdx.x < s) sm[threadIdx.x] = fmaxf(sm[threadIdx.x], sm[threadIdx.x + s]);
        __syncthreads();
    }
    if (threadIdx.x == 0) *dmax = sm[0];
}

// ---------------------------------------------------------------------------
// Kernel D: fused mask + leaky_relu + softmax + (att @ Hp), single adj pass.
// One wave owns 16 output rows x all 128 features (8 f16 WMMA acc tiles).
// m_i = max(0, src_i + dmax) is an upper bound on the row max -> p <= 1,
// no online rescaling needed. Row sums accumulated in fp32.
// dst[] (32 KB) is staged in LDS via async-to-LDS so the streaming VMEM path
// is dedicated to adj (the 256 MB HBM-bound operand).
// ---------------------------------------------------------------------------
__global__ __launch_bounds__(64)
void gat_attn(const float* __restrict__ adj, const _Float16* __restrict__ HpT,
              const float* __restrict__ src, const float* __restrict__ dst,
              const float* __restrict__ dmaxp, float* __restrict__ out) {
    __shared__ float sdst[N_NODES];            // 32 KB
    const int tid   = threadIdx.x;
    const int lane  = tid & 31;
    const int wave  = tid >> 5;
    const int gw    = blockIdx.x * 2 + wave;   // 512 waves total
    const int ibase = gw * 16;
    const int half  = lane >> 4;
    const int l16   = lane & 15;
    const int row   = ibase + l16;
    const int koff1 = half * 8;                // lanes<16: K 0..7 ; lanes>=16: K 8..15
    const int koff2 = koff1 + 16;              // + K 16..23 / 24..31

    // stage dst vector into LDS (async DMA path when available)
#if USE_ASYNC_LDS
#pragma unroll 4
    for (int i = tid * 4; i < N_NODES; i += 64 * 4) {
        __builtin_amdgcn_global_load_async_to_lds_b128(GASV(dst + i),
                                                       LASV(sdst + i), 0, 0);
    }
    __builtin_amdgcn_s_wait_asynccnt(0);
    __syncthreads();
#else
    for (int i = tid * 4; i < N_NODES; i += 64 * 4) {
        *(fx4*)(sdst + i) = *(const fx4*)(dst + i);
    }
    __syncthreads();
#endif

    const float dmax = *dmaxp;
    const float srow = src[row];
    const float m    = fmaxf(0.0f, srow + dmax);

    v8f acc[8] = {};
    float lsum = 0.0f;

    const float* adjrow = adj + (size_t)row * N_NODES;

    for (int j0 = 0; j0 < N_NODES; j0 += 32) {
        __builtin_prefetch((const void*)(adjrow + j0 + 256), 0, 0);

        fx4 av[4], dv[4];
        av[0] = *(const fx4*)(adjrow + j0 + koff1);
        av[1] = *(const fx4*)(adjrow + j0 + koff1 + 4);
        av[2] = *(const fx4*)(adjrow + j0 + koff2);
        av[3] = *(const fx4*)(adjrow + j0 + koff2 + 4);
        dv[0] = *(const fx4*)(sdst + j0 + koff1);
        dv[1] = *(const fx4*)(sdst + j0 + koff1 + 4);
        dv[2] = *(const fx4*)(sdst + j0 + koff2);
        dv[3] = *(const fx4*)(sdst + j0 + koff2 + 4);

        // A 16x32 f16 layout: this lane holds K pairs in order
        // {koff1..koff1+7, koff2..koff2+7}
        v16h Amat;
#pragma unroll
        for (int q = 0; q < 4; ++q) {
#pragma unroll
            for (int i = 0; i < 4; ++i) {
                const float s = srow + dv[q][i];
                const float e = (av[q][i] != 0.0f) ? ((s > 0.0f) ? s : 0.2f * s)
                                                   : 0.0f;
                const float p = __expf(e - m);
                lsum += p;
                Amat[q * 4 + i] = (_Float16)p;
            }
        }

#pragma unroll
        for (int t = 0; t < 8; ++t) {
            // B 32x16 f16: lane = column (feature), K {koff1..koff1+7, koff2..koff2+7} in HpT
            const _Float16* bp = HpT + (size_t)(t * 16 + l16) * N_NODES
                                 + j0;
            v8h blo = *(const v8h*)(bp + koff1);
            v8h bhi = *(const v8h*)(bp + koff2);
            v16h B;
#pragma unroll
            for (int i = 0; i < 8; ++i) { B[i] = blo[i]; B[i + 8] = bhi[i]; }
            acc[t] = __builtin_amdgcn_wmma_f32_16x16x32_f16(false, Amat, false, B,
                                                            (short)0, acc[t],
                                                            false, false);
        }
    }

    // combine the two K-half partial row sums: lanes l and l+16 share a row
    lsum += __shfl_xor(lsum, 16, 32);
    const float inv = 1.0f / lsum;             // lane holds inv for row l16

    // broadcast into C layout: VGPR k -> row k (lanes<16) / k+8 (lanes>=16)
    float invs[8];
#pragma unroll
    for (int k = 0; k < 8; ++k) invs[k] = __shfl(inv, k + half * 8, 32);

#pragma unroll
    for (int t = 0; t < 8; ++t) {
#pragma unroll
        for (int k = 0; k < 8; ++k) {
            out[(size_t)(ibase + k + 8 * half) * F_OUT + t * 16 + l16] =
                acc[t][k] * invs[k];
        }
    }
}

// ---------------------------------------------------------------------------
extern "C" void kernel_launch(void* const* d_in, const int* in_sizes, int n_in,
                              void* d_out, int out_size, void* d_ws, size_t ws_size,
                              hipStream_t stream) {
    const float* H   = (const float*)d_in[0];   // [8192, 768]
    const float* adj = (const float*)d_in[1];   // [8192, 8192]
    const float* W   = (const float*)d_in[2];   // [768, 128]
    const float* a   = (const float*)d_in[3];   // [256, 1]
    float* out = (float*)d_out;                 // [8192, 128]

    char* ws = (char*)d_ws;
    float*    Hp   = (float*)ws;                                       // 4 MB
    _Float16* HpT  = (_Float16*)(ws + (size_t)N_NODES * F_OUT * 4);    // 2 MB
    float*    src  = (float*)(ws + (size_t)6 * 1024 * 1024);           // 32 KB
    float*    dst  = src + N_NODES;                                    // 32 KB
    float*    dmax = dst + N_NODES;                                    // 4 B

    gat_hp_gemm<<<N_NODES / 16, 256, 0, stream>>>(H, W, Hp, HpT);
    gat_src_dst<<<N_NODES, 128, 0, stream>>>(Hp, a, src, dst);
    gat_dmax<<<1, 256, 0, stream>>>(dst, dmax);
    gat_attn<<<N_NODES / 32, 64, 0, stream>>>(adj, HpT, src, dst, dmax, out);
}
